// GenerativeModule_36661840839819
// MI455X (gfx1250) — matrix operand order repaired and checked by reading, not compiled, not run on hardware
//
#include <hip/hip_runtime.h>
#include <hip/hip_bf16.h>
#include <math.h>

#define L_LAYERS 6
#define NH 12
#define D_IN 256
#define DM 768
#define HD 64
#define BB 8
#define SS 1000
#define MROWS (BB*SS)   // 8000 rows, divisible by 16
#define SP 1024         // padded seq stride for V^T (keeps b128 loads in-bounds)

typedef __attribute__((ext_vector_type(16))) __bf16 bf16x16;
typedef __attribute__((ext_vector_type(8)))  __bf16 bf16x8;
typedef __attribute__((ext_vector_type(4)))  __bf16 bf16x4;
typedef __attribute__((ext_vector_type(8)))  float  f32x8;

// Native fptrunc (RNE) -> hardware bf16 convert on gfx1250.
__device__ inline __bf16 f2bf(float f) { return (__bf16)f; }

__device__ inline f32x8 wmma_bf16(bf16x16 a, bf16x16 b, f32x8 c) {
  return __builtin_amdgcn_wmma_f32_16x16x32_bf16(false, a, false, b, (short)0, c, false, false);
}

// ---------------------------------------------------------------------------
// Weight transpose + f32->bf16 convert: W[K,N] -> Wt[N,K].  LDS-tiled so both
// the global read and the global write are coalesced; run once per launch so
// every GEMM B-operand becomes two contiguous b128 loads.
// ---------------------------------------------------------------------------
__global__ __launch_bounds__(256) void transpose_cvt_kernel(
    const float* __restrict__ W, __bf16* __restrict__ Wt, int K, int N)
{
  __shared__ __bf16 tile[32][33];            // pad to dodge bank conflicts
  const int n0 = blockIdx.x * 32;
  const int k0 = blockIdx.y * 32;
  #pragma unroll
  for (int j = 0; j < 4; ++j) {
    const int k = k0 + threadIdx.y + j*8;
    tile[threadIdx.y + j*8][threadIdx.x] = f2bf(W[(size_t)k * N + n0 + threadIdx.x]);
  }
  __syncthreads();
  #pragma unroll
  for (int j = 0; j < 4; ++j) {
    const int n = n0 + threadIdx.y + j*8;
    Wt[(size_t)n * K + k0 + threadIdx.x] = tile[threadIdx.x][threadIdx.y + j*8];
  }
}

// ---------------------------------------------------------------------------
// GEMM, one wave computes a 16(M) x 64(N) tile: one shared A operand feeds 4
// WMMAs per 32-wide K step.  A[M,K] bf16 row-major, Bt[N,K] bf16 (transposed
// weights) -> both operands are 2 x b128 loads.
// A layout: lane = M row (L%16); elem e <- K = k0+16h+e.
// B layout: lane = N col; elem e <- K = k0+16h+e.
// C layout: lane holds col n=L%16, rows m = r + 8*(L/16).
// ---------------------------------------------------------------------------
enum { EPI_INPROJ=0, EPI_BF16=1, EPI_GELU=2, EPI_RESID=3, EPI_QKV=4, EPI_OUT=5 };

template<int EPI>
__global__ __launch_bounds__(32) void gemm_wmma_kernel(
    const __bf16* __restrict__ A, const __bf16* __restrict__ Bt,
    const float* __restrict__ bias, int K, int N,
    float* __restrict__ f32_out, const float* __restrict__ pos,
    __bf16* __restrict__ bf_out,
    __bf16* __restrict__ qh, __bf16* __restrict__ kh, __bf16* __restrict__ vt)
{
  const int lane = threadIdx.x & 31;
  const int ncol = lane & 15;
  const int half = lane >> 4;
  const int n0   = blockIdx.x * 64;
  const int m0   = blockIdx.y * 16;

  f32x8 acc[4] = {{},{},{},{}};
  const __bf16* ap = A + (size_t)(m0 + ncol) * K;
  const __bf16* bp0 = Bt + (size_t)(n0 +  0 + ncol) * K + 16*half;
  const __bf16* bp1 = Bt + (size_t)(n0 + 16 + ncol) * K + 16*half;
  const __bf16* bp2 = Bt + (size_t)(n0 + 32 + ncol) * K + 16*half;
  const __bf16* bp3 = Bt + (size_t)(n0 + 48 + ncol) * K + 16*half;

  for (int k0 = 0; k0 < K; k0 += 32) {
    bf16x8 alo = *(const bf16x8*)(ap + k0 + 16*half);
    bf16x8 ahi = *(const bf16x8*)(ap + k0 + 16*half + 8);
    bf16x16 av;
    #pragma unroll
    for (int j = 0; j < 8; ++j) { av[j] = alo[j]; av[8+j] = ahi[j]; }

    const __bf16* bps[4] = {bp0, bp1, bp2, bp3};
    #pragma unroll
    for (int t = 0; t < 4; ++t) {
      bf16x8 blo = *(const bf16x8*)(bps[t] + k0);
      bf16x8 bhi = *(const bf16x8*)(bps[t] + k0 + 8);
      bf16x16 bv;
      #pragma unroll
      for (int j = 0; j < 8; ++j) { bv[j] = blo[j]; bv[8+j] = bhi[j]; }
      acc[t] = wmma_bf16(av, bv, acc[t]);
    }
  }

  #pragma unroll
  for (int t = 0; t < 4; ++t) {
    const int n  = n0 + t*16 + ncol;
    const float bn = bias[n];
    #pragma unroll
    for (int r = 0; r < 8; ++r) {
      const int m = m0 + r + 8*half;
      float v = acc[t][r] + bn;
      if constexpr (EPI == EPI_INPROJ) {
        f32_out[(size_t)m * N + n] = v + pos[(size_t)(m % SS) * N + n];
      } else if constexpr (EPI == EPI_BF16) {
        bf_out[(size_t)m * N + n] = f2bf(v);
      } else if constexpr (EPI == EPI_GELU) {
        float g = 0.5f * v * (1.0f + erff(v * 0.70710678118f));   // exact GELU
        bf_out[(size_t)m * N + n] = f2bf(g);
      } else if constexpr (EPI == EPI_RESID) {
        f32_out[(size_t)m * N + n] += v;
      } else if constexpr (EPI == EPI_OUT) {
        f32_out[(size_t)m * N + n] = v;
      } else if constexpr (EPI == EPI_QKV) {
        const int b = m / SS, s = m - b * SS;
        const int part = n / DM;
        const int c2   = n - part * DM;
        const int head = c2 / HD;
        const int d    = c2 - head * HD;
        const size_t bh = (size_t)(b * NH + head);
        const __bf16 bvv = f2bf(v);
        if      (part == 0) qh[(bh * SS + s) * HD + d] = bvv;
        else if (part == 1) kh[(bh * SS + s) * HD + d] = bvv;
        else                vt[(bh * HD + d) * SP + s] = bvv;   // V transposed
      }
    }
  }
}

// ---------------------------------------------------------------------------
// LayerNorm: one wave per row (768 cols = 6 float4/lane), shfl reduction,
// writes bf16 rows (packed b64 stores) ready to be a WMMA A operand.
// ---------------------------------------------------------------------------
__global__ __launch_bounds__(256) void layernorm_kernel(
    const float* __restrict__ h, const float* __restrict__ g,
    const float* __restrict__ b, __bf16* __restrict__ out)
{
  const int lane = threadIdx.x & 31;
  const int wave = threadIdx.x >> 5;
  const int row  = blockIdx.x * 8 + wave;
  const float* hp = h + (size_t)row * DM;
  float4 vals[6];
  float s = 0.f, s2 = 0.f;
  #pragma unroll
  for (int i = 0; i < 6; ++i) {
    vals[i] = *(const float4*)(hp + i*128 + lane*4);
    s  += vals[i].x + vals[i].y + vals[i].z + vals[i].w;
    s2 += vals[i].x*vals[i].x + vals[i].y*vals[i].y
        + vals[i].z*vals[i].z + vals[i].w*vals[i].w;
  }
  #pragma unroll
  for (int m = 16; m >= 1; m >>= 1) {
    s  += __shfl_xor(s,  m, 32);
    s2 += __shfl_xor(s2, m, 32);
  }
  const float mu  = s * (1.0f/DM);
  const float var = s2 * (1.0f/DM) - mu*mu;
  const float rs  = rsqrtf(var + 1e-5f);
  __bf16* op = out + (size_t)row * DM;
  #pragma unroll
  for (int i = 0; i < 6; ++i) {
    const int c = i*128 + lane*4;
    const float xs[4] = {vals[i].x, vals[i].y, vals[i].z, vals[i].w};
    bf16x4 o4;
    #pragma unroll
    for (int j = 0; j < 4; ++j)
      o4[j] = f2bf((xs[j]-mu)*rs*g[c+j] + b[c+j]);
    *(bf16x4*)(op + c) = o4;
  }
}

// ---------------------------------------------------------------------------
// Flash attention, one wave per (b, head, 16-query tile).
// Scores computed TRANSPOSED: S^T = K_tile(A) x Q^T(B), so the C layout
// (lane = query col, keys striped over VGPRs) is exactly the A-operand layout
// for P x V  -> no cross-lane shuffle of probabilities needed.
// Online softmax: key dim is in-lane; one shfl_xor(16) merges the halves.
// ---------------------------------------------------------------------------
__global__ __launch_bounds__(32) void attn_kernel(
    const __bf16* __restrict__ qh, const __bf16* __restrict__ kh,
    const __bf16* __restrict__ vt, __bf16* __restrict__ o_out)
{
  const int lane = threadIdx.x & 31;
  const int col  = lane & 15;
  const int half = lane >> 4;
  const int qt   = blockIdx.x;              // 0..62
  const int bh   = blockIdx.y;              // 0..95
  const int bq   = bh / NH, hq = bh - bq * NH;

  const int q0 = qt * 16;
  const int qg = q0 + col;
  const int qc = qg < SS ? qg : SS - 1;

  // Q as B operand: elem e <- d = 32c + 16h + e (two b128 loads per 32-chunk)
  const __bf16* qp = qh + ((size_t)bh * SS + qc) * HD;
  bf16x16 qb[2];
  #pragma unroll
  for (int c = 0; c < 2; ++c) {
    bf16x8 lo = *(const bf16x8*)(qp + 32*c + 16*half);
    bf16x8 hi = *(const bf16x8*)(qp + 32*c + 16*half + 8);
    #pragma unroll
    for (int j = 0; j < 8; ++j) { qb[c][j] = lo[j]; qb[c][8+j] = hi[j]; }
  }

  f32x8 oacc[4] = {{},{},{},{}};
  float mi = -1e30f, li = 0.f;

  const __bf16* kbase = kh + (size_t)bh * SS * HD;
  const __bf16* vbase = vt + (size_t)bh * HD * SP;

  for (int kb = 0; kb < 32; ++kb) {          // 32 blocks of 32 keys = 1024 >= 1000
    bf16x16 ka[2][2];                        // K tiles as A operand
    #pragma unroll
    for (int t = 0; t < 2; ++t) {
      const int key = kb*32 + t*16 + col;
      const int kc  = key < SS ? key : SS - 1;
      const __bf16* kp = kbase + (size_t)kc * HD;
      #pragma unroll
      for (int c = 0; c < 2; ++c) {
        bf16x8 lo = *(const bf16x8*)(kp + 32*c + 16*half);
        bf16x8 hi = *(const bf16x8*)(kp + 32*c + 16*half + 8);
        #pragma unroll
        for (int j = 0; j < 8; ++j) { ka[t][c][j] = lo[j]; ka[t][c][8+j] = hi[j]; }
      }
    }
    f32x8 st0 = {}, st1 = {};
    st0 = wmma_bf16(ka[0][0], qb[0], st0);
    st0 = wmma_bf16(ka[0][1], qb[1], st0);
    st1 = wmma_bf16(ka[1][0], qb[0], st1);
    st1 = wmma_bf16(ka[1][1], qb[1], st1);

    float p0[8], p1[8], mx = -1e30f;
    #pragma unroll
    for (int r = 0; r < 8; ++r) {
      const int k0i = kb*32 + r + 8*half;
      const float a0 = (k0i      < SS) ? st0[r] * 0.125f : -1e30f;  // scale = HD^-1/2
      const float a1 = (k0i + 16 < SS) ? st1[r] * 0.125f : -1e30f;
      p0[r] = a0; p1[r] = a1;
      mx = fmaxf(mx, fmaxf(a0, a1));
    }
    mx = fmaxf(mx, __shfl_xor(mx, 16, 32));
    const float mn    = fmaxf(mi, mx);
    const float alpha = __expf(mi - mn);
    float rs = 0.f;
    #pragma unroll
    for (int r = 0; r < 8; ++r) {
      p0[r] = __expf(p0[r] - mn);
      p1[r] = __expf(p1[r] - mn);
      rs += p0[r] + p1[r];
    }
    rs += __shfl_xor(rs, 16, 32);
    li = li * alpha + rs;
    mi = mn;

    // Probabilities are already in A-operand layout: pack to bf16.
    bf16x16 pa;
    #pragma unroll
    for (int j = 0; j < 8; ++j) { pa[j] = f2bf(p0[j]); pa[8+j] = f2bf(p1[j]); }

    float ar[8];
    #pragma unroll
    for (int r = 0; r < 8; ++r) ar[r] = __shfl(alpha, r + 8*half, 32);

    #pragma unroll
    for (int t = 0; t < 4; ++t) {            // 4 d-tiles of 16
      const __bf16* vp = vbase + (size_t)(t*16 + col) * SP + kb*32 + 8*half;
      bf16x8 lo = *(const bf16x8*)(vp);
      bf16x8 hi = *(const bf16x8*)(vp + 16);
      bf16x16 vb;
      #pragma unroll
      for (int j = 0; j < 8; ++j) { vb[j] = lo[j]; vb[8+j] = hi[j]; }
      #pragma unroll
      for (int r = 0; r < 8; ++r) oacc[t][r] *= ar[r];
      oacc[t] = wmma_bf16(pa, vb, oacc[t]);
    }
  }

  float lr[8];
  #pragma unroll
  for (int r = 0; r < 8; ++r) lr[r] = __shfl(li, r + 8*half, 32);
  #pragma unroll
  for (int r = 0; r < 8; ++r) {
    const int q = q0 + r + 8*half;
    if (q < SS) {
      const size_t m  = (size_t)bq * SS + q;
      const float inv = 1.0f / lr[r];
      #pragma unroll
      for (int t = 0; t < 4; ++t)
        o_out[m * DM + hq*HD + t*16 + col] = f2bf(oacc[t][r] * inv);
    }
  }
}

// Vectorized f32 -> bf16 converter (n divisible by 4).
__global__ void cvt_f32_bf16(const float* __restrict__ in,
                             __bf16* __restrict__ out, int n4) {
  const int i = blockIdx.x * blockDim.x + threadIdx.x;
  if (i < n4) {
    float4 v = *(const float4*)(in + (size_t)i*4);
    bf16x4 o4 = { f2bf(v.x), f2bf(v.y), f2bf(v.z), f2bf(v.w) };
    *(bf16x4*)(out + (size_t)i*4) = o4;
  }
}

// ---------------------------------------------------------------------------
extern "C" void kernel_launch(void* const* d_in, const int* in_sizes, int n_in,
                              void* d_out, int out_size, void* d_ws, size_t ws_size,
                              hipStream_t stream)
{
  (void)in_sizes; (void)n_in; (void)out_size; (void)ws_size;
  const float* x    = (const float*)d_in[0];
  const float* pos  = (const float*)d_in[1];
  const float* Win  = (const float*)d_in[2];
  const float* b_in = (const float*)d_in[3];
  const float* ln_g = (const float*)d_in[4];
  const float* ln_b = (const float*)d_in[5];
  const float* Wqkv = (const float*)d_in[6];
  const float* bqkv = (const float*)d_in[7];
  const float* Wproj= (const float*)d_in[8];
  const float* bproj= (const float*)d_in[9];
  const float* W1   = (const float*)d_in[10];
  const float* b1   = (const float*)d_in[11];
  const float* W2   = (const float*)d_in[12];
  const float* b2   = (const float*)d_in[13];
  const float* Wout = (const float*)d_in[14];
  const float* bout = (const float*)d_in[15];
  float* out = (float*)d_out;

  char* wsp = (char*)d_ws;
  auto alloc = [&](size_t bytes) -> char* {
    char* p = wsp; wsp += (bytes + 255) & ~(size_t)255; return p;
  };
  float*  h   = (float*) alloc((size_t)MROWS * DM * 4);       // residual stream
  __bf16* hn  = (__bf16*)alloc((size_t)MROWS * DM * 2);       // LN output
  __bf16* qhb = (__bf16*)alloc((size_t)BB*NH*SS*HD * 2);      // Q per-head
  __bf16* khb = (__bf16*)alloc((size_t)BB*NH*SS*HD * 2);      // K per-head
  __bf16* vtb = (__bf16*)alloc((size_t)BB*NH*HD*SP * 2);      // V^T per-head (padded)
  __bf16* ob  = (__bf16*)alloc((size_t)MROWS * DM * 2);       // attn out
  __bf16* pb  = (__bf16*)alloc((size_t)MROWS * DM * 2);       // proj out
  __bf16* gb  = (__bf16*)alloc((size_t)MROWS * 4*DM * 2);     // gelu(ffn1)
  __bf16* xb  = (__bf16*)alloc((size_t)MROWS * D_IN * 2);     // x as bf16
  __bf16* hb  = (__bf16*)alloc((size_t)MROWS * DM * 2);       // final h as bf16
  // transposed bf16 weights
  __bf16* WinT  = (__bf16*)alloc((size_t)D_IN * DM * 2);
  __bf16* WqkvT = (__bf16*)alloc((size_t)L_LAYERS * DM * 3*DM * 2);
  __bf16* WprojT= (__bf16*)alloc((size_t)L_LAYERS * DM * DM * 2);
  __bf16* W1T   = (__bf16*)alloc((size_t)L_LAYERS * DM * 4*DM * 2);
  __bf16* W2T   = (__bf16*)alloc((size_t)L_LAYERS * 4*DM * DM * 2);
  __bf16* WoutT = (__bf16*)alloc((size_t)DM * D_IN * 2);

  const dim3 tb(32, 8);
  transpose_cvt_kernel<<<dim3(DM/32, D_IN/32), tb, 0, stream>>>(Win, WinT, D_IN, DM);
  transpose_cvt_kernel<<<dim3(D_IN/32, DM/32), tb, 0, stream>>>(Wout, WoutT, DM, D_IN);
  for (int i = 0; i < L_LAYERS; ++i) {
    transpose_cvt_kernel<<<dim3(3*DM/32, DM/32), tb, 0, stream>>>(
        Wqkv + (size_t)i*DM*3*DM, WqkvT + (size_t)i*DM*3*DM, DM, 3*DM);
    transpose_cvt_kernel<<<dim3(DM/32, DM/32), tb, 0, stream>>>(
        Wproj + (size_t)i*DM*DM, WprojT + (size_t)i*DM*DM, DM, DM);
    transpose_cvt_kernel<<<dim3(4*DM/32, DM/32), tb, 0, stream>>>(
        W1 + (size_t)i*DM*4*DM, W1T + (size_t)i*DM*4*DM, DM, 4*DM);
    transpose_cvt_kernel<<<dim3(DM/32, 4*DM/32), tb, 0, stream>>>(
        W2 + (size_t)i*4*DM*DM, W2T + (size_t)i*4*DM*DM, 4*DM, DM);
  }

  { const int n4 = MROWS * D_IN / 4;
    cvt_f32_bf16<<<(n4 + 255)/256, 256, 0, stream>>>(x, xb, n4); }

  gemm_wmma_kernel<EPI_INPROJ><<<dim3(DM/64, MROWS/16), 32, 0, stream>>>(
      xb, WinT, b_in, D_IN, DM, h, pos, nullptr, nullptr, nullptr, nullptr);

  for (int i = 0; i < L_LAYERS; ++i) {
    layernorm_kernel<<<SS, 256, 0, stream>>>(h, ln_g + i*DM, ln_b + i*DM, hn);

    gemm_wmma_kernel<EPI_QKV><<<dim3(3*DM/64, MROWS/16), 32, 0, stream>>>(
        hn, WqkvT + (size_t)i*DM*3*DM, bqkv + (size_t)i*3*DM, DM, 3*DM,
        nullptr, nullptr, nullptr, qhb, khb, vtb);

    attn_kernel<<<dim3(63, BB*NH), 32, 0, stream>>>(qhb, khb, vtb, ob);

    gemm_wmma_kernel<EPI_BF16><<<dim3(DM/64, MROWS/16), 32, 0, stream>>>(
        ob, WprojT + (size_t)i*DM*DM, bproj + (size_t)i*DM, DM, DM,
        nullptr, nullptr, pb, nullptr, nullptr, nullptr);

    gemm_wmma_kernel<EPI_GELU><<<dim3(4*DM/64, MROWS/16), 32, 0, stream>>>(
        pb, W1T + (size_t)i*DM*4*DM, b1 + (size_t)i*4*DM, DM, 4*DM,
        nullptr, nullptr, gb, nullptr, nullptr, nullptr);

    gemm_wmma_kernel<EPI_RESID><<<dim3(DM/64, MROWS/16), 32, 0, stream>>>(
        gb, W2T + (size_t)i*4*DM*DM, b2 + (size_t)i*DM, 4*DM, DM,
        h, nullptr, nullptr, nullptr, nullptr, nullptr);
  }

  { const int n4 = MROWS * DM / 4;
    cvt_f32_bf16<<<(n4 + 255)/256, 256, 0, stream>>>(h, hb, n4); }

  gemm_wmma_kernel<EPI_OUT><<<dim3(D_IN/64, MROWS/16), 32, 0, stream>>>(
      hb, WoutT, bout, DM, D_IN, out, nullptr, nullptr, nullptr, nullptr, nullptr);
}
